// BlockSparseMoE_40072044871689
// MI455X (gfx1250) — matrix operand order repaired and checked by reading, not compiled, not run on hardware
//
#include <hip/hip_runtime.h>

// ---------------------------------------------------------------------------
// MoE (top-2 of 8) for MI455X / gfx1250, wave32, bf16 WMMA path.
// Sparse gather per expert. Gathered x and intermediate activations are kept
// in WMMA A-fragment tile layout in global memory, so both GEMMs' A-staging
// is a pure async copy (global_load_async_to_lds_b128 + s_wait_asynccnt).
// Weights are fp32->bf16 converted (v_cvt_pk_bf16_f32) while staging to LDS.
// Loop order per K-step: global loads -> ds frag loads + WMMA -> cvt/ds
// stores -> wait/barrier, so WMMA never waits on staging stores (DScnt is
// in-order; stores are issued after the fragment loads WMMA depends on).
// ---------------------------------------------------------------------------

constexpr int T = 2048;
constexpr int H = 1024;
constexpr int F = 3584;
constexpr int E = 8;

constexpr int BM = 64;      // block rows (token slots)
constexpr int BN = 128;     // block cols
constexpr int BK = 32;      // K step (WMMA bf16 K)
constexpr int HT = H / 32;  // xg k-tiles per 16-slot row (32)
constexpr int FT = F / 32;  // act k-tiles per 16-slot row (112)

typedef __attribute__((ext_vector_type(16))) __bf16 v16bf;
typedef __attribute__((ext_vector_type(8)))  float  v8f;

union Frag { v16bf v; uint4 q[2]; unsigned short u[16]; };

__device__ __forceinline__ unsigned short f2bf(float f) {
    union { __bf16 h; unsigned short u; } r;
    r.h = (__bf16)f;
    return r.u;
}

// two fp32 -> packed bf16 pair (v_cvt_pk_bf16_f32)
__device__ __forceinline__ unsigned cvt2(float a, float b) {
#if __has_builtin(__builtin_amdgcn_cvt_pk_bf16_f32)
    unsigned out;
    auto r = __builtin_amdgcn_cvt_pk_bf16_f32(a, b);
    __builtin_memcpy(&out, &r, 4);
    return out;
#else
    union { __bf16 h[2]; unsigned u; } r;
    r.h[0] = (__bf16)a;
    r.h[1] = (__bf16)b;
    return r.u;
#endif
}

__device__ __forceinline__ unsigned long long pack4(float a, float b, float c, float d) {
    return (unsigned long long)cvt2(a, b) | ((unsigned long long)cvt2(c, d) << 32);
}

__device__ __forceinline__ v16bf ldfrag(const unsigned short* p) {
    Frag f;
    f.q[0] = ((const uint4*)p)[0];   // ds_load_b128
    f.q[1] = ((const uint4*)p)[1];   // ds_load_b128
    return f.v;
}

#define WMMA_BF16(A, B, C) \
    __builtin_amdgcn_wmma_f32_16x16x32_bf16(false, (A), false, (B), (short)0, (C), false, false)

// ---------------------------------------------------------------------------
// Router: logits -> top-2 -> renormalized weights -> per-expert token lists
// ---------------------------------------------------------------------------
__global__ __launch_bounds__(256) void moe_router(
    const float* __restrict__ x, const float* __restrict__ gate,
    int* __restrict__ cnt, int* __restrict__ toks, float* __restrict__ wts)
{
    __shared__ float gs[E * H];                       // 32 KB
    const int tid = threadIdx.x;
    for (int i = tid; i < E * H; i += 256) gs[i] = gate[i];
    __syncthreads();

    const int t = blockIdx.x * 256 + tid;
    const float* xr = x + (size_t)t * H;
    float acc[E];
#pragma unroll
    for (int e = 0; e < E; ++e) acc[e] = 0.f;
    for (int k = 0; k < H; ++k) {
        float xv = xr[k];
#pragma unroll
        for (int e = 0; e < E; ++e) acc[e] += xv * gs[e * H + k];
    }
    float v1 = -3.0e38f, v2 = -3.0e38f;
    int i1 = 0, i2 = 0;
#pragma unroll
    for (int e = 0; e < E; ++e) {
        float a = acc[e];
        if (a > v1)      { v2 = v1; i2 = i1; v1 = a; i1 = e; }
        else if (a > v2) { v2 = a;  i2 = e; }
    }
    float p2  = __expf(v2 - v1);                 // softmax denoms cancel in top-2
    float inv = __builtin_amdgcn_rcpf(1.f + p2);
    int s1 = atomicAdd(&cnt[i1], 1);
    toks[i1 * T + s1] = t; wts[i1 * T + s1] = inv;
    int s2 = atomicAdd(&cnt[i2], 1);
    toks[i2 * T + s2] = t; wts[i2 * T + s2] = p2 * inv;
}

// ---------------------------------------------------------------------------
// Gather: xg[slot] = bf16(x[tok[slot]]) in A-fragment tile layout.
// ---------------------------------------------------------------------------
__global__ __launch_bounds__(256) void moe_gather(
    const float* __restrict__ x, const int* __restrict__ cnt,
    const int* __restrict__ toks, unsigned short* __restrict__ xg, int e)
{
    const int count = cnt[e];
    const int base  = blockIdx.x * 64;
    if (base >= count) return;
    const int tid = threadIdx.x;
    const int m   = base + (tid >> 2);
    if (m >= count) return;
    const int tok = toks[e * T + m];
    const float* src = x + (size_t)tok * H;
    const int mt = m >> 4, mres = m & 15;
#pragma unroll 8
    for (int it = 0; it < 64; ++it) {
        int k4 = ((tid & 3) * 64 + it) * 4;     // 0..1020, step 4
        float4 v = *(const float4*)(src + k4);
        int h  = (k4 >> 3) & 1;
        int e0 = (k4 & 7) | ((k4 & 16) >> 1);
        int kt = k4 >> 5;
        size_t idx = (((size_t)mt * HT + kt) * 32 + (mres + 16 * h)) * 16 + e0;
        *(unsigned long long*)&xg[idx] = pack4(v.x, v.y, v.z, v.w);
    }
}

// ---------------------------------------------------------------------------
// GEMM1 + SiLU*u fused.  Block 64x128, K-step 32 over H, double-buffered LDS
// in fragment layout.  8 waves (2Mx4N): 2 A frags x 2 N tiles x {g,u}
// -> 8 WMMA per K step.
// ---------------------------------------------------------------------------
__global__ __launch_bounds__(256) void moe_gemm1(
    const unsigned short* __restrict__ xg, const float* __restrict__ w13,
    const int* __restrict__ cnt, unsigned short* __restrict__ act, int e)
{
    __shared__ __align__(16) unsigned short xsF[2][4][32][16];   //  4 KB x2
    __shared__ __align__(16) unsigned short wgF[2][8][32][16];   //  8 KB x2
    __shared__ __align__(16) unsigned short wuF[2][8][32][16];   //  8 KB x2

    const int count = cnt[e];
    const int m0 = blockIdx.y * BM;
    if (m0 >= count) return;
    const int n0 = blockIdx.x * BN;
    const float* w13e = w13 + (size_t)e * (2 * F) * H;

    const int tid  = threadIdx.x;
    const int lane = tid & 31;
    const int wave = tid >> 5;
    const int wr = wave >> 2;          // 0..1 : pair of M tiles
    const int wc = wave & 3;           // 0..3 : 32-col strip

    // ---- A-copy: one 16B tile-chunk per thread (4 tiles x 64 uint4) ----
    const int mtb = blockIdx.y * 4 + (tid >> 6);
    const unsigned long long abase =
        (unsigned long long)(uintptr_t)((const uint4*)xg + (size_t)mtb * HT * 64 + (tid & 63));
    const unsigned xsF_lds = (unsigned)(uintptr_t)&xsF[0][0][0][0] + (unsigned)tid * 16u;
    constexpr unsigned XSF_BUFSZ = 4u * 32u * 16u * 2u;          // bytes per buffer

    // ---- loop-invariant weight staging addresses ----
    const float* gsrc[4]; const float* usrc[4]; unsigned wdst[4];
#pragma unroll
    for (int jj = 0; jj < 4; ++jj) {
        int j = tid + 256 * jj;
        int n = j >> 3, k4 = (j & 7) * 4;
        gsrc[jj] = w13e + (size_t)(n0 + n) * H + k4;
        usrc[jj] = w13e + (size_t)(F + n0 + n) * H + k4;
        wdst[jj] = (unsigned)(((n >> 4) * 32 + (n & 15) + 16 * ((k4 >> 3) & 1)) * 16 + ((k4 & 7) | ((k4 & 16) >> 1)));
    }

    v8f cg[2][2] = {{v8f{}, v8f{}}, {v8f{}, v8f{}}};
    v8f cu[2][2] = {{v8f{}, v8f{}}, {v8f{}, v8f{}}};

    float4 rg[4], ru[4];                 // staged weight registers

    auto stage_load = [&](int k0) {
        {   // x tiles: async global -> LDS (other buffer), 16B per lane
            unsigned long long src = abase + (unsigned long long)(k0 >> 5) * 1024ull;
            unsigned dst = xsF_lds + (unsigned)(((k0 >> 5) & 1)) * XSF_BUFSZ;
            asm volatile("global_load_async_to_lds_b128 %0, %1, off"
                         :: "v"(dst), "v"(src) : "memory");
        }
        const bool pf = (k0 + 2 * BK) < H;
#pragma unroll
        for (int jj = 0; jj < 4; ++jj) {
            rg[jj] = *(const float4*)(gsrc[jj] + k0);
            ru[jj] = *(const float4*)(usrc[jj] + k0);
            if (pf) {
                __builtin_prefetch(gsrc[jj] + k0 + 2 * BK, 0, 3);    // global_prefetch_b8
                __builtin_prefetch(usrc[jj] + k0 + 2 * BK, 0, 3);
            }
        }
    };
    auto stage_store = [&](int buf) {
        unsigned short* gb = &wgF[buf][0][0][0];
        unsigned short* ub = &wuF[buf][0][0][0];
#pragma unroll
        for (int jj = 0; jj < 4; ++jj) {
            unsigned g01 = cvt2(rg[jj].x, rg[jj].y), g23 = cvt2(rg[jj].z, rg[jj].w);
            unsigned u01 = cvt2(ru[jj].x, ru[jj].y), u23 = cvt2(ru[jj].z, ru[jj].w);
            unsigned short* pg = gb + wdst[jj];
            unsigned short* pu = ub + wdst[jj];
            pg[0]  = (unsigned short)g01; pg[1] = (unsigned short)(g01 >> 16);
            pg[2] = (unsigned short)g23; pg[3] = (unsigned short)(g23 >> 16);
            pu[0]  = (unsigned short)u01; pu[1] = (unsigned short)(u01 >> 16);
            pu[2] = (unsigned short)u23; pu[3] = (unsigned short)(u23 >> 16);
        }
    };

    stage_load(0);
    stage_store(0);
    asm volatile("s_wait_asynccnt 0x0" ::: "memory");
    __syncthreads();
    constexpr int NK = H / BK;
    for (int kt = 0; kt < NK; ++kt) {
        const int cur  = kt & 1;
        const bool more = (kt + 1) < NK;
        if (more) stage_load((kt + 1) * BK);            // LOADcnt/ASYNCcnt only
        v16bf a0 = ldfrag(&xsF[cur][wr * 2 + 0][lane][0]);
        v16bf a1 = ldfrag(&xsF[cur][wr * 2 + 1][lane][0]);
#pragma unroll
        for (int tt = 0; tt < 2; ++tt) {                // DScnt: frag loads only
            v16bf bg = ldfrag(&wgF[cur][wc * 2 + tt][lane][0]);
            cg[0][tt] = WMMA_BF16(a0, bg, cg[0][tt]);
            cg[1][tt] = WMMA_BF16(a1, bg, cg[1][tt]);
            v16bf bu = ldfrag(&wuF[cur][wc * 2 + tt][lane][0]);
            cu[0][tt] = WMMA_BF16(a0, bu, cu[0][tt]);
            cu[1][tt] = WMMA_BF16(a1, bu, cu[1][tt]);
        }
        if (more) stage_store(cur ^ 1);                 // co-executes with WMMA
        asm volatile("s_wait_asynccnt 0x0" ::: "memory");
        __syncthreads();
    }

    // ---- silu(g)*u, store bf16 into A-fragment-layout act tensor ----
    const int hD  = lane >> 4;
    const int l16 = lane & 15;
#pragma unroll
    for (int mm = 0; mm < 2; ++mm) {
#pragma unroll
        for (int tt = 0; tt < 2; ++tt) {
            int f  = n0 + wc * 32 + tt * 16 + l16;
            int kt = f >> 5;
            int hk = (f >> 3) & 1;
            int ee = (f & 7) | ((f & 16) >> 1);
#pragma unroll
            for (int r = 0; r < 8; ++r) {
                int m = m0 + (wr * 2 + mm) * 16 + hD * 8 + r;
                if (m < count) {
                    float g = cg[mm][tt][r], u = cu[mm][tt][r];
                    float s = g * __builtin_amdgcn_rcpf(1.f + __expf(-g));
                    size_t idx = ((((size_t)(m >> 4)) * FT + kt) * 32 +
                                  ((m & 15) + 16 * hk)) * 16 + ee;
                    act[idx] = f2bf(s * u);
                }
            }
        }
    }
}

// ---------------------------------------------------------------------------
// GEMM2: y = act @ w2[e]^T.  A staging = async LDS copy.  Scaled scatter-add
// into out via f32 hardware atomics.
// ---------------------------------------------------------------------------
__global__ __launch_bounds__(256) void moe_gemm2(
    const unsigned short* __restrict__ act, const float* __restrict__ w2,
    const int* __restrict__ cnt, const int* __restrict__ toks,
    const float* __restrict__ wts, float* __restrict__ out, int e)
{
    __shared__ __align__(16) unsigned short asF[2][4][32][16];   //  4 KB x2
    __shared__ __align__(16) unsigned short wbF[2][8][32][16];   //  8 KB x2

    const int count = cnt[e];
    const int m0 = blockIdx.y * BM;
    if (m0 >= count) return;
    const int n0 = blockIdx.x * BN;
    const float* w2e = w2 + (size_t)e * H * F;

    const int tid  = threadIdx.x;
    const int lane = tid & 31;
    const int wave = tid >> 5;
    const int wr = wave >> 2;
    const int wc = wave & 3;

    const int mt = blockIdx.y * 4 + (tid >> 6);
    const unsigned long long abase =
        (unsigned long long)(uintptr_t)((const uint4*)act + (size_t)mt * FT * 64 + (tid & 63));
    const unsigned asF_lds = (unsigned)(uintptr_t)&asF[0][0][0][0] + (unsigned)tid * 16u;
    constexpr unsigned ASF_BUFSZ = 4u * 32u * 16u * 2u;          // bytes per buffer

    const float* bsrc[4]; unsigned wdst[4];
#pragma unroll
    for (int jj = 0; jj < 4; ++jj) {
        int j = tid + 256 * jj;
        int n = j >> 3, k4 = (j & 7) * 4;
        bsrc[jj] = w2e + (size_t)(n0 + n) * F + k4;
        wdst[jj] = (unsigned)(((n >> 4) * 32 + (n & 15) + 16 * ((k4 >> 3) & 1)) * 16 + ((k4 & 7) | ((k4 & 16) >> 1)));
    }

    v8f c[2][2] = {{v8f{}, v8f{}}, {v8f{}, v8f{}}};

    float4 rb[4];

    auto stage_load = [&](int k0) {
        {
            unsigned long long src = abase + (unsigned long long)(k0 >> 5) * 1024ull;
            unsigned dst = asF_lds + (unsigned)(((k0 >> 5) & 1)) * ASF_BUFSZ;
            asm volatile("global_load_async_to_lds_b128 %0, %1, off"
                         :: "v"(dst), "v"(src) : "memory");
        }
        const bool pf = (k0 + 2 * BK) < F;
#pragma unroll
        for (int jj = 0; jj < 4; ++jj) {
            rb[jj] = *(const float4*)(bsrc[jj] + k0);
            if (pf) __builtin_prefetch(bsrc[jj] + k0 + 2 * BK, 0, 3);
        }
    };
    auto stage_store = [&](int buf) {
        unsigned short* bb = &wbF[buf][0][0][0];
#pragma unroll
        for (int jj = 0; jj < 4; ++jj) {
            unsigned q01 = cvt2(rb[jj].x, rb[jj].y), q23 = cvt2(rb[jj].z, rb[jj].w);
            unsigned short* p = bb + wdst[jj];
            p[0]  = (unsigned short)q01; p[1] = (unsigned short)(q01 >> 16);
            p[2] = (unsigned short)q23; p[3] = (unsigned short)(q23 >> 16);
        }
    };

    stage_load(0);
    stage_store(0);
    asm volatile("s_wait_asynccnt 0x0" ::: "memory");
    __syncthreads();
    constexpr int NK = F / BK;
    for (int kt = 0; kt < NK; ++kt) {
        const int cur  = kt & 1;
        const bool more = (kt + 1) < NK;
        if (more) stage_load((kt + 1) * BK);
        v16bf a0 = ldfrag(&asF[cur][wr * 2 + 0][lane][0]);
        v16bf a1 = ldfrag(&asF[cur][wr * 2 + 1][lane][0]);
#pragma unroll
        for (int tt = 0; tt < 2; ++tt) {
            v16bf b = ldfrag(&wbF[cur][wc * 2 + tt][lane][0]);
            c[0][tt] = WMMA_BF16(a0, b, c[0][tt]);
            c[1][tt] = WMMA_BF16(a1, b, c[1][tt]);
        }
        if (more) stage_store(cur ^ 1);
        asm volatile("s_wait_asynccnt 0x0" ::: "memory");
        __syncthreads();
    }

    const int hD  = lane >> 4;
    const int l16 = lane & 15;
#pragma unroll
    for (int mm = 0; mm < 2; ++mm) {
#pragma unroll
        for (int tt = 0; tt < 2; ++tt) {
            int col = n0 + wc * 32 + tt * 16 + l16;
#pragma unroll
            for (int r = 0; r < 8; ++r) {
                int m = m0 + (wr * 2 + mm) * 16 + hD * 8 + r;
                if (m < count) {
                    int   tok = toks[e * T + m];
                    float w   = wts[e * T + m];
                    unsafeAtomicAdd(&out[(size_t)tok * H + col], w * c[mm][tt][r]);
                }
            }
        }
    }
}

// ---------------------------------------------------------------------------
// Launch: memsets -> router -> per-expert (gather ; gemm1 ; gemm2).
// Workspace: [cnt 256B | toks E*T i32 | wts E*T f32 | xg 4MB | act 14.7MB]
// ---------------------------------------------------------------------------
extern "C" void kernel_launch(void* const* d_in, const int* in_sizes, int n_in,
                              void* d_out, int out_size, void* d_ws, size_t ws_size,
                              hipStream_t stream) {
    (void)in_sizes; (void)n_in; (void)ws_size;
    const float* x      = (const float*)d_in[0];
    const float* gate_w = (const float*)d_in[1];
    const float* w13    = (const float*)d_in[2];
    const float* w2     = (const float*)d_in[3];
    float* out = (float*)d_out;

    char* ws = (char*)d_ws;
    int*            cnt  = (int*)ws;
    int*            toks = (int*)(ws + 256);
    float*          wts  = (float*)(ws + 256 + (size_t)E * T * sizeof(int));
    unsigned short* xg   = (unsigned short*)(ws + 256 + 2 * (size_t)E * T * sizeof(int));
    unsigned short* act  = (unsigned short*)((char*)xg + (size_t)T * H * sizeof(unsigned short));

    hipMemsetAsync(cnt, 0, 256, stream);
    hipMemsetAsync(out, 0, (size_t)out_size * sizeof(float), stream);

    moe_router<<<T / 256, 256, 0, stream>>>(x, gate_w, cnt, toks, wts);

    for (int e = 0; e < E; ++e) {
        moe_gather<<<T / 64, 256, 0, stream>>>(x, cnt, toks, xg, e);
        moe_gemm1<<<dim3(F / BN, T / BM), 256, 0, stream>>>(xg, w13, cnt, act, e);
        moe_gemm2<<<dim3(H / BN, T / BM), 256, 0, stream>>>(act, w2, cnt, toks, wts, out, e);
    }
}
